// Attention_40750649704906
// MI455X (gfx1250) — matrix operand order repaired and checked by reading, not compiled, not run on hardware
//
#include <hip/hip_runtime.h>

// ---------------- constants ----------------
constexpr int Bb = 2;
constexpr int Tt = 2048;
constexpr int Dd = 2048;
constexpr int Hh = 16;
constexpr int Gg = 4;
constexpr int Kk = 128;
constexpr float SCALE_F = 0.08838834764831845f; // 128^-0.5
constexpr float NEG_BIG = -1e30f;

typedef __attribute__((ext_vector_type(16))) __bf16 v16bf;
typedef __attribute__((ext_vector_type(8)))  float  v8f;

union Frag {
  v16bf v;
  unsigned short u[16];
  uint4 q[2];   // two 16-byte halves -> ds_load_b128 / global_load_b128
};

__device__ __forceinline__ unsigned short f2bf(float f) {
  unsigned int u = __float_as_uint(f);
  unsigned int r = 0x7FFFu + ((u >> 16) & 1u);
  return (unsigned short)((u + r) >> 16);
}

// ---------------- f32 -> bf16 convert ----------------
__global__ void f2bf_kernel(const float* __restrict__ s, unsigned short* __restrict__ d, int n) {
  int i = blockIdx.x * blockDim.x + threadIdx.x;
  if (i < n) d[i] = f2bf(s[i]);
}

// f32 (rows x cols, row-major) -> bf16 transposed (cols x rows). cols = 1<<csh.
__global__ void f2bf_t_kernel(const float* __restrict__ s, unsigned short* __restrict__ d,
                              int rows, int csh) {
  int i = blockIdx.x * blockDim.x + threadIdx.x;
  int cols = 1 << csh;
  if (i < (rows << csh)) {
    int r = i >> csh;
    int c = i & (cols - 1);           // coalesced read along c
    d[(size_t)c * rows + r] = f2bf(s[i]);
  }
}

// ---------------- bf16 WMMA GEMM: C(MxN) = A(MxK) * Bt(NxK)^T ----------------
// A row-major MxK; Bt row-major NxK (pre-transposed weights).
// M % 64 == 0, N % 256 == 0, K % 32 == 0. 256 threads = 8 waves (2x4),
// block tile 64x256; each wave 32x64 = 2x4 WMMA accumulators.
// Single-buffered LDS; scalar staging pointers (no arrays) so every staging
// access lowers to global_load_b128 (not flat) and the compiler can pipeline.
__global__ __launch_bounds__(256) void gemm_bf16_kernel(
    const unsigned short* __restrict__ A, const unsigned short* __restrict__ Bt,
    float* __restrict__ C, int M, int N, int Kd) {
  __shared__ alignas(16) unsigned short As[64][48];    // 64 x 32 (+pad)
  __shared__ alignas(16) unsigned short Bs[256][48];   // 256 x 32 (+pad)

  const int tid  = threadIdx.x;
  const int w    = tid >> 5;
  const int lane = tid & 31;
  const int wm = w >> 2, wn = w & 3;
  const int m0 = blockIdx.y * 64;
  const int n0 = blockIdx.x * 256;
  const int m16  = lane & 15;
  const int kh8  = (lane >> 4) * 8;    // A-fragment interleaved K-half
  const int kb16 = (lane >> 4) * 16;   // unused; B fragments use kh8 like A

  // hoisted per-thread staging addresses (all scalars -> global addrspace inferred)
  const int ar = tid >> 2, ac8 = (tid & 3) * 8;
  const int bn = tid >> 2, bc = (tid & 3) * 8;   // B chunk row/col for j = tid + k*256: row += 64 per k
  const unsigned short* aG  = A + (size_t)(m0 + ar) * Kd + ac8;
  const unsigned short* bG0 = Bt + (size_t)(n0 + bn) * Kd + bc;
  const unsigned short* bG1 = Bt + (size_t)(n0 + bn + 64) * Kd + bc;
  const unsigned short* bG2 = Bt + (size_t)(n0 + bn + 128) * Kd + bc;
  const unsigned short* bG3 = Bt + (size_t)(n0 + bn + 192) * Kd + bc;
  unsigned short* aL  = &As[ar][ac8];
  unsigned short* bL0 = &Bs[bn][bc];
  unsigned short* bL1 = &Bs[bn + 64][bc];
  unsigned short* bL2 = &Bs[bn + 128][bc];
  unsigned short* bL3 = &Bs[bn + 192][bc];

  v8f acc[2][4];
#pragma unroll
  for (int i = 0; i < 2; ++i)
#pragma unroll
    for (int j = 0; j < 4; ++j)
#pragma unroll
      for (int r = 0; r < 8; ++r) acc[i][j][r] = 0.0f;

  for (int kt = 0; kt < Kd; kt += 32) {
    *(uint4*)aL  = *(const uint4*)aG;  aG  += 32;
    *(uint4*)bL0 = *(const uint4*)bG0; bG0 += 32;
    *(uint4*)bL1 = *(const uint4*)bG1; bG1 += 32;
    *(uint4*)bL2 = *(const uint4*)bG2; bG2 += 32;
    *(uint4*)bL3 = *(const uint4*)bG3; bG3 += 32;
    __syncthreads();

    Frag af[2], bf[4];
#pragma unroll
    for (int im = 0; im < 2; ++im) {
      int mr = wm * 32 + im * 16 + m16;
      af[im].q[0] = *(const uint4*)&As[mr][kh8];
      af[im].q[1] = *(const uint4*)&As[mr][16 + kh8];
    }
#pragma unroll
    for (int in = 0; in < 4; ++in) {
      int nc = wn * 64 + in * 16 + m16;
      bf[in].q[0] = *(const uint4*)&Bs[nc][kh8];
      bf[in].q[1] = *(const uint4*)&Bs[nc][16 + kh8];
    }
#pragma unroll
    for (int im = 0; im < 2; ++im)
#pragma unroll
      for (int in = 0; in < 4; ++in)
        acc[im][in] = __builtin_amdgcn_wmma_f32_16x16x32_bf16(
            false, af[im].v, false, bf[in].v, (short)0, acc[im][in], false, false);
    __syncthreads();
  }

#pragma unroll
  for (int im = 0; im < 2; ++im)
#pragma unroll
    for (int in = 0; in < 4; ++in)
#pragma unroll
      for (int r = 0; r < 8; ++r) {
        int m = m0 + wm * 32 + im * 16 + r + 8 * (lane >> 4);
        int n = n0 + wn * 64 + in * 16 + m16;
        C[(size_t)m * N + n] = acc[im][in][r];
      }
}

// ---------------- RMS-norm + RoPE, f32 -> bf16 [b, head, t, 128] ----------------
__global__ __launch_bounds__(128) void qknorm_rope_kernel(
    const float* __restrict__ src, int row_stride, int head_stride,
    const float* __restrict__ nw, const float* __restrict__ cosb,
    const float* __restrict__ sinb, unsigned short* __restrict__ dst, int nheads) {
  int h = blockIdx.x, t = blockIdx.y, b = blockIdx.z;
  int c = threadIdx.x;  // 0..127
  const float* row = src + (size_t)(b * Tt + t) * row_stride + h * head_stride;
  float x = row[c];
  float ss = x * x;
  for (int m = 1; m < 32; m <<= 1) ss += __shfl_xor(ss, m);
  __shared__ float red[4];
  if ((threadIdx.x & 31) == 0) red[threadIdx.x >> 5] = ss;
  __syncthreads();
  float tot = red[0] + red[1] + red[2] + red[3];
  float inv = rsqrtf(tot * (1.0f / 128.0f) + 1e-6f);
  float xn = x * inv * nw[c];
  int pc = (c < 64) ? c + 64 : c - 64;
  float xp = row[pc] * inv * nw[pc];
  float rh = (c < 64) ? -xp : xp;
  float cs = cosb[(size_t)(b * Tt + t) * Kk + c];
  float sn = sinb[(size_t)(b * Tt + t) * Kk + c];
  dst[(((size_t)(b * nheads + h)) * Tt + t) * Kk + c] = f2bf(xn * cs + rh * sn);
}

// ---------------- V repack: [b,t,g,c] f32 -> [b,g,c,t] bf16 (transposed) --------
__global__ void vpack_kernel(const float* __restrict__ vbuf, unsigned short* __restrict__ dst) {
  int i = blockIdx.x * blockDim.x + threadIdx.x;  // B*T*G*K = 2097152
  int c = i & 127;
  int g = (i >> 7) & 3;
  int t = (i >> 9) & 2047;
  int b = i >> 20;
  dst[(((size_t)(b * Gg + g)) * Kk + c) * Tt + t] = f2bf(vbuf[i]);
}

// ---------------- flash attention + sigmoid gate, bf16 WMMA ----------------
// grid (T/128, H, B), block 256 = 8 waves. Wave w owns q rows [t0+16w, t0+16w+16).
// kh: [b,g,t,c] bf16 ; vt: [b,g,c,t] bf16 (keys along fastest dim for PV B-frags)
__global__ __launch_bounds__(256) void attn_kernel(
    const unsigned short* __restrict__ qh, const unsigned short* __restrict__ kh,
    const unsigned short* __restrict__ vt, const float* __restrict__ qgate,
    const int* __restrict__ seg, unsigned short* __restrict__ outb) {
  __shared__ alignas(16) unsigned short Ks[32][144];   // keys x headdim (+pad)
  __shared__ alignas(16) unsigned short Vs[128][48];   // headdim x keys (+pad)
  __shared__ alignas(16) unsigned short Ps[8][16][24]; // per-wave P staging
  __shared__ int segk[32];

  const int tt = blockIdx.x, h = blockIdx.y, b = blockIdx.z;
  const int t0 = tt * 128;
  const int tid = threadIdx.x, w = tid >> 5, lane = tid & 31;
  const int g = h >> 2;  // h / n_rep
  const int m16  = lane & 15;
  const int kh8  = (lane >> 4) * 8;
  const int kb16 = (lane >> 4) * 16;
  const int qrow0 = t0 + w * 16;

  // Q fragments (A layout), resident: 4 chunks of K=32, b128 global loads
  Frag qf[4];
  {
    const unsigned short* qr = qh + (((size_t)(b * Hh + h)) * Tt + (qrow0 + m16)) * Kk;
#pragma unroll
    for (int kc = 0; kc < 4; ++kc) {
      qf[kc].q[0] = *(const uint4*)(qr + kc * 32 + kh8);
      qf[kc].q[1] = *(const uint4*)(qr + kc * 32 + 16 + kh8);
    }
  }
  int posq[8], segq[8];
#pragma unroll
  for (int r = 0; r < 8; ++r) {
    int tq = qrow0 + r + 8 * (lane >> 4);
    posq[r] = tq;
    segq[r] = seg[b * Tt + tq];
  }
  float mr[8], lr[8];
  v8f o[8];
#pragma unroll
  for (int r = 0; r < 8; ++r) { mr[r] = NEG_BIG; lr[r] = 0.0f; }
#pragma unroll
  for (int oc = 0; oc < 8; ++oc)
#pragma unroll
    for (int r = 0; r < 8; ++r) o[oc][r] = 0.0f;

  // hoisted per-thread tile-load pointers (advance by constants per s-step)
  const size_t kvbase = ((size_t)(b * Gg + g));
  const int kr0 = tid >> 4,        kc0 = (tid & 15) * 8;
  const int kr1 = (tid >> 4) + 16, kc1 = kc0;
  const unsigned short* kG0 = kh + (kvbase * Tt + kr0) * Kk + kc0;
  const unsigned short* kG1 = kh + (kvbase * Tt + kr1) * Kk + kc1;
  const int vc0 = tid >> 2,        vk0 = (tid & 3) * 8;
  const int vc1 = (tid >> 2) + 64, vk1 = vk0;
  const unsigned short* vG0 = vt + (kvbase * Kk + vc0) * Tt + vk0;
  const unsigned short* vG1 = vt + (kvbase * Kk + vc1) * Tt + vk1;
  unsigned short* kL0 = &Ks[kr0][kc0];
  unsigned short* kL1 = &Ks[kr1][kc1];
  unsigned short* vL0 = &Vs[vc0][vk0];
  unsigned short* vL1 = &Vs[vc1][vk1];
  const int* segG = seg + b * Tt + (tid & 31);

  const int s_end = t0 + 128;
  for (int st = 0; st < s_end; st += 32) {
    *(uint4*)kL0 = *(const uint4*)kG0;
    *(uint4*)kL1 = *(const uint4*)kG1;
    *(uint4*)vL0 = *(const uint4*)vG0;
    *(uint4*)vL1 = *(const uint4*)vG1;
    kG0 += 32 * Kk; kG1 += 32 * Kk;
    vG0 += 32;      vG1 += 32;
    if (tid < 32) segk[tid] = segG[0];
    segG += 32;
    __syncthreads();

    // logits S (16 x 32) = Q (16x128) . K_tile^T
    v8f sc[2];
#pragma unroll
    for (int in = 0; in < 2; ++in) {
#pragma unroll
      for (int r = 0; r < 8; ++r) sc[in][r] = 0.0f;
      const int krow = in * 16 + m16;
#pragma unroll
      for (int kc = 0; kc < 4; ++kc) {
        Frag bf;  // B layout: B[kk][n] = Ks[krow][kc*32 + kk], interleaved K-halves
        bf.q[0] = *(const uint4*)&Ks[krow][kc * 32 + kh8];
        bf.q[1] = *(const uint4*)&Ks[krow][kc * 32 + 16 + kh8];
        sc[in] = __builtin_amdgcn_wmma_f32_16x16x32_bf16(
            false, qf[kc].v, false, bf.v, (short)0, sc[in], false, false);
      }
    }
    // scale + causal/segment mask
#pragma unroll
    for (int in = 0; in < 2; ++in) {
      int sidx = st + in * 16 + m16;
      int sgk = segk[in * 16 + m16];
#pragma unroll
      for (int r = 0; r < 8; ++r) {
        float val = sc[in][r] * SCALE_F;
        bool ok = (posq[r] >= sidx) && (segq[r] == sgk);
        sc[in][r] = ok ? val : NEG_BIG;
      }
    }
    // online softmax per row
#pragma unroll
    for (int r = 0; r < 8; ++r) {
      float v = fmaxf(sc[0][r], sc[1][r]);
      v = fmaxf(v, __shfl_xor(v, 1));
      v = fmaxf(v, __shfl_xor(v, 2));
      v = fmaxf(v, __shfl_xor(v, 4));
      v = fmaxf(v, __shfl_xor(v, 8));
      float mnew = fmaxf(mr[r], v);
      float alpha = __expf(mr[r] - mnew);
      mr[r] = mnew;
      float e0 = __expf(sc[0][r] - mnew);
      float e1 = __expf(sc[1][r] - mnew);
      sc[0][r] = e0; sc[1][r] = e1;
      float rs = e0 + e1;
      rs += __shfl_xor(rs, 1);
      rs += __shfl_xor(rs, 2);
      rs += __shfl_xor(rs, 4);
      rs += __shfl_xor(rs, 8);
      lr[r] = lr[r] * alpha + rs;
#pragma unroll
      for (int oc = 0; oc < 8; ++oc) o[oc][r] *= alpha;
    }
    // C layout -> A layout for P via per-wave LDS staging
#pragma unroll
    for (int r = 0; r < 8; ++r) {
      int m = r + 8 * (lane >> 4);
      Ps[w][m][m16]      = f2bf(sc[0][r]);
      Ps[w][m][16 + m16] = f2bf(sc[1][r]);
    }
    asm volatile("s_wait_dscnt 0" ::: "memory");
    Frag pa;
    pa.q[0] = *(const uint4*)&Ps[w][m16][kh8];
    pa.q[1] = *(const uint4*)&Ps[w][m16][16 + kh8];
    // O += P (16x32) . V_tile (32x128)
#pragma unroll
    for (int oc = 0; oc < 8; ++oc) {
      Frag vf;  // B layout: B[kk][n] = Vs[oc*16 + n][kk], interleaved K-halves
      vf.q[0] = *(const uint4*)&Vs[oc * 16 + m16][kh8];
      vf.q[1] = *(const uint4*)&Vs[oc * 16 + m16][16 + kh8];
      o[oc] = __builtin_amdgcn_wmma_f32_16x16x32_bf16(
          false, pa.v, false, vf.v, (short)0, o[oc], false, false);
    }
    __syncthreads();
  }

  // epilogue: 1/l, sigmoid gate, store bf16 [b,t, h*128 + c]
#pragma unroll
  for (int r = 0; r < 8; ++r) {
    int tq = qrow0 + r + 8 * (lane >> 4);
    float inv = 1.0f / lr[r];
#pragma unroll
    for (int oc = 0; oc < 8; ++oc) {
      int c = oc * 16 + m16;
      float gv = qgate[((size_t)(b * Tt + tq)) * (2 * Hh * Kk) + h * 256 + 128 + c];
      float sig = 1.0f / (1.0f + __expf(-gv));
      outb[((size_t)(b * Tt + tq)) * (Hh * Kk) + h * Kk + c] = f2bf(o[oc][r] * inv * sig);
    }
  }
}

// ---------------- host launcher ----------------
extern "C" void kernel_launch(void* const* d_in, const int* in_sizes, int n_in,
                              void* d_out, int out_size, void* d_ws, size_t ws_size,
                              hipStream_t stream) {
  const float* hidden = (const float*)d_in[0];
  const float* cosb   = (const float*)d_in[1];
  const float* sinb   = (const float*)d_in[2];
  const int*   segids = (const int*)d_in[3];
  // d_in[4] position_ids: arange(T), implicit
  const float* Wq = (const float*)d_in[5];
  const float* Wk = (const float*)d_in[6];
  const float* Wv = (const float*)d_in[7];
  const float* Wo = (const float*)d_in[8];
  const float* qnw = (const float*)d_in[9];
  const float* knw = (const float*)d_in[10];
  float* out = (float*)d_out;

  const int M = Bb * Tt;  // 4096

  char* ws = (char*)d_ws;
  size_t off = 0;
  auto take = [&](size_t bytes) -> char* {
    char* p = ws + off;
    off += (bytes + 255) & ~(size_t)255;
    return p;
  };
  unsigned short* hid_bf = (unsigned short*)take((size_t)M * Dd * 2);
  unsigned short* wq_bt  = (unsigned short*)take((size_t)Dd * (2 * Hh * Kk) * 2);  // (2HK) x D
  unsigned short* wk_bt  = (unsigned short*)take((size_t)Dd * (Gg * Kk) * 2);      // (GK) x D
  unsigned short* wv_bt  = (unsigned short*)take((size_t)Dd * (Gg * Kk) * 2);
  unsigned short* wo_bt  = (unsigned short*)take((size_t)(Hh * Kk) * Dd * 2);      // D x (HK)
  float* qgate = (float*)take((size_t)M * (2 * Hh * Kk) * 4);
  float* kbuf  = (float*)take((size_t)M * (Gg * Kk) * 4);
  float* vbuf  = (float*)take((size_t)M * (Gg * Kk) * 4);
  unsigned short* qh_bf  = (unsigned short*)take((size_t)Bb * Hh * Tt * Kk * 2);
  unsigned short* kh_bf  = (unsigned short*)take((size_t)Bb * Gg * Tt * Kk * 2);
  unsigned short* vt_bf  = (unsigned short*)take((size_t)Bb * Gg * Tt * Kk * 2);
  unsigned short* attn_bf = (unsigned short*)take((size_t)M * (Hh * Kk) * 2);

  // 1) convert operands to bf16 (weights transposed to N x K for the GEMM B path)
  f2bf_kernel<<<(M * Dd + 255) / 256, 256, 0, stream>>>(hidden, hid_bf, M * Dd);
  f2bf_t_kernel<<<(Dd * 2 * Hh * Kk + 255) / 256, 256, 0, stream>>>(Wq, wq_bt, Dd, 12); // cols=4096
  f2bf_t_kernel<<<(Dd * Gg * Kk + 255) / 256, 256, 0, stream>>>(Wk, wk_bt, Dd, 9);      // cols=512
  f2bf_t_kernel<<<(Dd * Gg * Kk + 255) / 256, 256, 0, stream>>>(Wv, wv_bt, Dd, 9);
  f2bf_t_kernel<<<(Hh * Kk * Dd + 255) / 256, 256, 0, stream>>>(Wo, wo_bt, Hh * Kk, 11); // cols=2048

  // 2) projections (WMMA GEMMs), block tile 64x256
  gemm_bf16_kernel<<<dim3((2 * Hh * Kk) / 256, M / 64), 256, 0, stream>>>(
      hid_bf, wq_bt, qgate, M, 2 * Hh * Kk, Dd);
  gemm_bf16_kernel<<<dim3((Gg * Kk) / 256, M / 64), 256, 0, stream>>>(
      hid_bf, wk_bt, kbuf, M, Gg * Kk, Dd);
  gemm_bf16_kernel<<<dim3((Gg * Kk) / 256, M / 64), 256, 0, stream>>>(
      hid_bf, wv_bt, vbuf, M, Gg * Kk, Dd);

  // 3) RMS-norm + RoPE into [b, head, t, 128] bf16; V repack to [b,g,c,t]
  qknorm_rope_kernel<<<dim3(Hh, Tt, Bb), 128, 0, stream>>>(
      qgate, 2 * Hh * Kk, 2 * Kk, qnw, cosb, sinb, qh_bf, Hh);
  qknorm_rope_kernel<<<dim3(Gg, Tt, Bb), 128, 0, stream>>>(
      kbuf, Gg * Kk, Kk, knw, cosb, sinb, kh_bf, Gg);
  vpack_kernel<<<(Bb * Tt * Gg * Kk) / 256, 256, 0, stream>>>(vbuf, vt_bf);

  // 4) flash attention + gate
  attn_kernel<<<dim3(Tt / 128, Hh, Bb), 256, 0, stream>>>(
      qh_bf, kh_bf, vt_bf, qgate, segids, attn_bf);

  // 5) output projection
  gemm_bf16_kernel<<<dim3(Dd / 256, M / 64), 256, 0, stream>>>(
      attn_bf, wo_bt, out, M, Dd, Hh * Kk);
}
